// QAttn_72980084293761
// MI455X (gfx1250) — matrix operand order repaired and checked by reading, not compiled, not run on hardware
//
#include <hip/hip_runtime.h>

// ---------------------------------------------------------------------------
// CDNA5 (gfx1250) quantized ViT attention block.
// wave32 + v_wmma_f32_16x16x32_bf16 + global_load_async_to_lds_b128 (GEMM staging)
// + TDM tensor_load_to_lds for attention K/Q tiles.
// B=64 N=197 C=768 H=12 D=64;  M = B*N = 12608 (multiple of 64)
// ---------------------------------------------------------------------------

typedef __bf16 bf16_t;
typedef __attribute__((ext_vector_type(8)))  bf16_t   v8bf;
typedef __attribute__((ext_vector_type(16))) bf16_t   v16bf;
typedef __attribute__((ext_vector_type(8)))  float    v8f;
typedef __attribute__((ext_vector_type(4)))  unsigned v4u;
typedef __attribute__((ext_vector_type(8)))  int      v8i;
typedef __attribute__((ext_vector_type(4)))  int      v4i;

#define SHUF16(a,b) __builtin_shufflevector(a,b,0,1,2,3,4,5,6,7,8,9,10,11,12,13,14,15)

__device__ __forceinline__ unsigned short f2bf(float f){
  unsigned u=__float_as_uint(f);
  unsigned r=((u>>16)&1u)+0x7FFFu;            // round-to-nearest-even
  return (unsigned short)((u+r)>>16);
}
__device__ __forceinline__ float bf2f(unsigned short h){
  return __uint_as_float(((unsigned)h)<<16);
}
// monotonic float <-> ordered-uint encoding for atomic min/max
__device__ __forceinline__ unsigned f2ord(float f){
  unsigned u=__float_as_uint(f);
  return (u&0x80000000u)? ~u : (u|0x80000000u);
}
__device__ __forceinline__ float ord2f(unsigned u){
  return __uint_as_float((u&0x80000000u)? (u&0x7FFFFFFFu) : ~u);
}
__device__ __forceinline__ void get_scale(const unsigned* scal,int slot,float& s,float& z){
  float mn=ord2f(scal[slot]), mx=ord2f(scal[slot+1]);
  s=fmaxf((mx-mn)*(1.0f/255.0f),1e-8f);
  z=rintf(-mn/s);
}
__device__ __forceinline__ float fq1(float x,float s,float z){
  float q=fminf(fmaxf(rintf(x/s)+z,0.0f),255.0f);
  return (q-z)*s;
}
__device__ __forceinline__ v8f wmma_bf16(v16bf a,v16bf b,v8f c){
  return __builtin_amdgcn_wmma_f32_16x16x32_bf16(false,a,false,b,(short)0,c,false,false);
}
// A fragment (16x32 bf16): lane holds row (lane&15); lanes<16: K0-7,16-23; lanes>=16: K8-15,24-31
__device__ __forceinline__ v16bf frag_a(const unsigned short* row,int lane){
  int ko=(lane&16)?8:0;
  v8bf a0=*(const v8bf*)(row+ko);
  v8bf a1=*(const v8bf*)(row+ko+16);
  return SHUF16(a0,a1);
}
// B fragment (32x16 bf16): lane holds column (lane&15); lanes<16: K0-7,16-23; lanes>=16: K8-15,24-31
__device__ __forceinline__ v16bf frag_b(const unsigned short* col,int lane){
  int ko=(lane&16)?8:0;
  v8bf b0=*(const v8bf*)(col+ko);
  v8bf b1=*(const v8bf*)(col+ko+16);
  return SHUF16(b0,b1);
}
// CDNA5 async memory->LDS (ASYNCcnt), ISA §08: per-lane LDS dest addr, 16B per lane
__device__ __forceinline__ void async_b128(const unsigned short* g, unsigned short* l){
  unsigned lo=(unsigned)(uintptr_t)l;                 // low 32 bits of LDS aperture addr
  unsigned long long ga=(unsigned long long)(uintptr_t)g;
  asm volatile("global_load_async_to_lds_b128 %0, %1, off" :: "v"(lo), "v"(ga) : "memory");
}
__device__ __forceinline__ void wait_async(){
  asm volatile("s_wait_asynccnt 0x0" ::: "memory");
}
// Tensor Data Mover: 2D bf16 tile (rowElems per row, `rows` rows, stride in elements)
// into LDS with a 4-DWORD pad every 32 DWORDs (i.e. 64-half rows -> 72-half LDS pitch).
// D# layout per ISA §8.3 (group0) / §8.4 (group1); groups 2/3 zero (2D tensor).
// This toolchain exposes the 6-arg builtin (g0, g1, g2, g3, extra group, cpol).
__device__ __forceinline__ void tdm_load_2d(const unsigned short* gaddr, unsigned lds_off,
                                            unsigned rowElems, unsigned rows,
                                            unsigned rowStrideElems){
  unsigned long long ga=(unsigned long long)(uintptr_t)gaddr;
  v4u g0;
  g0[0]=1u;                                            // count=1, user descriptor
  g0[1]=lds_off;                                       // lds_addr (bytes)
  g0[2]=(unsigned)ga;                                  // global_addr[31:0]
  g0[3]=(unsigned)((ga>>32)&0x1FFFFFFu)|(2u<<30);      // global_addr[56:32] | type=2
  v8i g1;
  g1[0]=(int)((1u<<16)|(1u<<20)|(4u<<22)|(3u<<25));    // data_size=2B, pad_en, 32dw/4dw pad
  g1[1]=(int)(rowElems<<16);                           // tensor_dim0[15:0] @ bits 63:48
  g1[2]=(int)(rows<<16);                               // tensor_dim1[15:0] @ bits 95:80
  g1[3]=(int)(rowElems<<16);                           // tile_dim0 @ bits 127:112
  g1[4]=(int)rows;                                     // tile_dim1 @ bits 143:128
  g1[5]=(int)rowStrideElems;                           // tensor_dim0_stride[31:0]
  g1[6]=0; g1[7]=0;
  v4i z4={0,0,0,0};
  v8i z8={0,0,0,0,0,0,0,0};
  __builtin_amdgcn_tensor_load_to_lds(g0,g1,z4,z4,z8,0);
}
__device__ __forceinline__ float2 block_minmax(float fmn,float fmx){
  __shared__ float rmn[256], rmx[256];
  int t=threadIdx.x;
  rmn[t]=fmn; rmx[t]=fmx; __syncthreads();
  for(int s=128;s>0;s>>=1){
    if(t<s){ rmn[t]=fminf(rmn[t],rmn[t+s]); rmx[t]=fmaxf(rmx[t],rmx[t+s]); }
    __syncthreads();
  }
  return make_float2(rmn[0],rmx[0]);
}

// --- scale-slot table in ws: [0,1] qkv  [2..7] q/k/v  [8,9] attn  [10,11] attn_out  [12,13] proj
__global__ void init_scal_kernel(unsigned* scal){
  int t=threadIdx.x;
  if(t<64) scal[t]=(t&1)?0u:0xFFFFFFFFu;   // even=min slot (atomicMin), odd=max slot (atomicMax)
}

// per-output-row symmetric int8 weight fake-quant -> bf16
__global__ __launch_bounds__(256) void quant_w_kernel(const float* __restrict__ w,
                                                      unsigned short* __restrict__ o,int K){
  __shared__ float ra[256];
  int row=blockIdx.x, t=threadIdx.x;
  const float* wr=w+(size_t)row*K;
  float amax=0.0f;
  for(int c=t;c<K;c+=256) amax=fmaxf(amax,fabsf(wr[c]));
  ra[t]=amax; __syncthreads();
  for(int s=128;s>0;s>>=1){ if(t<s) ra[t]=fmaxf(ra[t],ra[t+s]); __syncthreads(); }
  float scale=fmaxf(ra[0]*(1.0f/127.0f),1e-8f);
  unsigned short* orow=o+(size_t)row*K;
  for(int c=t;c<K;c+=256){
    float q=fminf(fmaxf(rintf(wr[c]/scale),-128.0f),127.0f);
    orow[c]=f2bf(q*scale);
  }
}

__global__ void x2bf_kernel(const float* __restrict__ x,unsigned short* __restrict__ o,long long n){
  long long stride=(long long)gridDim.x*blockDim.x;
  for(long long i=blockIdx.x*(long long)blockDim.x+threadIdx.x;i<n;i+=stride)
    o[i]=f2bf(x[i]);
}

// C[M,N] = A[M,K] * W[N,K]^T + bias
// block tile 64x128, 8 waves (4x2), each wave 4 16x16 WMMA tiles; async double-buffered LDS
__global__ __launch_bounds__(256) void gemm_bf16_kernel(
    const unsigned short* __restrict__ A, const unsigned short* __restrict__ W,
    const float* __restrict__ bias,
    unsigned short* __restrict__ Cb, float* __restrict__ Cf,
    unsigned* scal,int slotG,int segBase,int segW,int N,int K){
  __shared__ __attribute__((aligned(16))) unsigned short ldsA[2][64][40];
  __shared__ __attribute__((aligned(16))) unsigned short ldsW[2][128][40];
  int t=threadIdx.x, lane=t&31, w=t>>5;
  int wm=w&3, wn=(w>>2)*64;                 // wave col base within 128-wide block tile
  int bx=blockIdx.x, by=blockIdx.y;
  // per-thread async staging slots
  int ar=t>>2, ac=(t&3)*8;                  // A: 64x32 = 256 b128 chunks, 1/thread
  int wr=t>>1, wc=(t&1)*16;                 // W: 128x32 = 512 b128 chunks, 2/thread
  const unsigned short* Ag=A+(size_t)(by*64+ar)*K+ac;
  const unsigned short* Wg=W+(size_t)(bx*128+wr)*K+wc;
  auto load_tile=[&](int buf,int k0){
    async_b128(Ag+k0,      &ldsA[buf][ar][ac]);
    async_b128(Wg+k0,      &ldsW[buf][wr][wc]);
    async_b128(Wg+k0+8,    &ldsW[buf][wr][wc+8]);
  };
  v8f acc[4]={{},{},{},{}};
  load_tile(0,0);
  wait_async(); __syncthreads();
  int buf=0;
  for(int k0=0;k0<K;k0+=32){
    if(k0+32<K) load_tile(buf^1,k0+32);     // overlap next-tile DMA with WMMA
    // read all fragments first so ds_loads batch with partial dscnt waits
    v16bf af=frag_a(&ldsA[buf][wm*16+(lane&15)][0],lane);
    v16bf bfr[4];
#pragma unroll
    for(int j=0;j<4;j++)
      bfr[j]=frag_b(&ldsW[buf][wn+j*16+(lane&15)][0],lane);
#pragma unroll
    for(int j=0;j<4;j++)
      acc[j]=wmma_bf16(af,bfr[j],acc[j]);
    wait_async(); __syncthreads();
    buf^=1;
  }
  float fmn=3.0e38f, fmx=-3.0e38f;
  int rowb=by*64+wm*16+((lane&16)?8:0);
#pragma unroll
  for(int j=0;j<4;j++){
    int colg=bx*128+wn+j*16+(lane&15);
    float bb=bias[colg];
    for(int i=0;i<8;i++){
      float v=acc[j][i]+bb;
      fmn=fminf(fmn,v); fmx=fmaxf(fmx,v);
      size_t idx=(size_t)(rowb+i)*N+colg;
      if(Cb) Cb[idx]=f2bf(v);
      if(Cf) Cf[idx]=v;
    }
  }
  float2 mm=block_minmax(fmn,fmx);
  if(t==0){
    atomicMin(&scal[slotG],f2ord(mm.x));
    atomicMax(&scal[slotG+1],f2ord(mm.y));
    if(segW>0){
      int si=(bx*128)/segW;   // 768 % 128 == 0: whole block inside one segment
      atomicMin(&scal[segBase+2*si],f2ord(mm.x));
      atomicMax(&scal[segBase+2*si+1],f2ord(mm.y));
    }
  }
}

// in-place fake-quant of a [rows x cols] bf16 slab; optional post-quant min/max recording
__global__ __launch_bounds__(256) void fq_apply_kernel(
    unsigned short* __restrict__ buf,long long rows,int rowStride,int colOff,int cols,
    const unsigned* scal,int slot,unsigned* outSlots){
  float s,z; get_scale(scal,slot,s,z);
  long long total=rows*(long long)cols;
  long long stride=(long long)gridDim.x*blockDim.x;
  float fmn=3.0e38f, fmx=-3.0e38f;
  for(long long i=blockIdx.x*(long long)blockDim.x+threadIdx.x;i<total;i+=stride){
    long long r=i/cols; int c=(int)(i-r*cols);
    unsigned short* p=buf+r*rowStride+colOff+c;
    float v=fq1(bf2f(*p),s,z);
    *p=f2bf(v);
    fmn=fminf(fmn,v); fmx=fmaxf(fmx,v);
  }
  if(outSlots){
    float2 mm=block_minmax(fmn,fmx);
    if(threadIdx.x==0){
      atomicMin(&outSlots[0],f2ord(mm.x));
      atomicMax(&outSlots[1],f2ord(mm.y));
    }
  }
}

__global__ void fq_out_kernel(float* buf,long long n,const unsigned* scal,int slot){
  float s,z; get_scale(scal,slot,s,z);
  long long stride=(long long)gridDim.x*blockDim.x;
  for(long long i=blockIdx.x*(long long)blockDim.x+threadIdx.x;i<n;i+=stride)
    buf[i]=fq1(buf[i],s,z);
}

// Q/K staging via Tensor Data Mover. Wave 0 issues two TDM descriptors and drains
// TENSORcnt; pad rows (>=197) are zero-filled with ds stores to disjoint addresses.
__device__ __forceinline__ void stage_qk(const unsigned short* __restrict__ qkv,
                                         unsigned short (*sQ)[72], unsigned short (*sK)[72],
                                         int b,int h,int mt,int t){
  uint4 zz=make_uint4(0,0,0,0);
  for(int cid=t;cid<144;cid+=256){                       // sQ pad rows (16x9 chunks)
    int r=cid/9, c=(cid%9)*8;
    if(mt*16+r>=197) *(uint4*)&sQ[r][c]=zz;
  }
  for(int cid=t;cid<243;cid+=256){                       // sK pad rows 197..223
    int m=197+cid/9, c=(cid%9)*8;
    *(uint4*)&sK[m][c]=zz;
  }
  if(t<32){                                              // one wave drives the TDM
    unsigned qrows=(unsigned)((197-mt*16)<16?(197-mt*16):16);
    tdm_load_2d(qkv+(size_t)(b*197+mt*16)*2304+h*64,
                (unsigned)(uintptr_t)&sQ[0][0], 64, qrows, 2304);
    tdm_load_2d(qkv+(size_t)(b*197)*2304+768+h*64,
                (unsigned)(uintptr_t)&sK[0][0], 64, 197, 2304);
    __builtin_amdgcn_s_wait_tensorcnt((short)0);
  }
}

// attention pass 1: per (b,h,16-row tile) compute S = 0.125 * Q K^T via WMMA, reduce min/max
__global__ __launch_bounds__(256) void attn_pass1_kernel(const unsigned short* __restrict__ qkv,
                                                         unsigned* scal){
  __shared__ __attribute__((aligned(16))) unsigned short sQ[16][72];
  __shared__ __attribute__((aligned(16))) unsigned short sK[224][72];
  int t=threadIdx.x, lane=t&31, w=t>>5;
  int mt=blockIdx.x, h=blockIdx.y, b=blockIdx.z;
  stage_qk(qkv,sQ,sK,b,h,mt,t);
  __syncthreads();
  float fmn=3.0e38f, fmx=-3.0e38f;
  for(int tt=w;tt<14;tt+=8){                 // wave-uniform -> EXEC all-ones for WMMA
    v8f acc={};
    for(int kk=0;kk<2;kk++){
      v16bf af=frag_a(&sQ[lane&15][kk*32],lane);
      v16bf bfr=frag_b(&sK[tt*16+(lane&15)][kk*32],lane);
      acc=wmma_bf16(af,bfr,acc);
    }
    int m=tt*16+(lane&15);
    int nb=mt*16+((lane&16)?8:0);
    for(int i=0;i<8;i++){
      if(nb+i<197 && m<197){
        float v=acc[i]*0.125f;
        fmn=fminf(fmn,v); fmx=fmaxf(fmx,v);
      }
    }
  }
  float2 mm=block_minmax(fmn,fmx);
  if(t==0){ atomicMin(&scal[8],f2ord(mm.x)); atomicMax(&scal[9],f2ord(mm.y)); }
}

// attention pass 2: S recompute -> I-BERT integer softmax (scale from pass 1) -> P @ V
__global__ __launch_bounds__(256) void attn_pass2_kernel(const unsigned short* __restrict__ qkv,
                                                         unsigned short* __restrict__ AO,
                                                         unsigned* scal){
  __shared__ __attribute__((aligned(16))) unsigned short sQ[16][72];
  __shared__ __attribute__((aligned(16))) unsigned short sK[224][72];
  __shared__ __attribute__((aligned(16))) unsigned short sVt[64][232];  // V transposed [d][m]
  __shared__ __attribute__((aligned(16))) float sS[16][224];
  __shared__ __attribute__((aligned(16))) unsigned short sP[16][224];
  __shared__ float sred[16][17];
  int t=threadIdx.x, lane=t&31, w=t>>5;
  int mt=blockIdx.x, h=blockIdx.y, b=blockIdx.z;
  stage_qk(qkv,sQ,sK,b,h,mt,t);
  for(int cid=t;cid<1792;cid+=256){            // V loaded with manual transpose
    int m=cid>>3, dd=(cid&7)*8;
    uint4 val=make_uint4(0,0,0,0);
    if(m<197) val=*(const uint4*)(qkv+(size_t)(b*197+m)*2304+1536+h*64+dd);
    union{uint4 u; unsigned short s[8];} cv; cv.u=val;
    for(int j=0;j<8;j++) sVt[dd+j][m]=cv.s[j];
  }
  __syncthreads();
  for(int tt=w;tt<14;tt+=8){
    v8f acc={};
    for(int kk=0;kk<2;kk++){
      v16bf af=frag_a(&sQ[lane&15][kk*32],lane);
      v16bf bfr=frag_b(&sK[tt*16+(lane&15)][kk*32],lane);
      acc=wmma_bf16(af,bfr,acc);
    }
    int m=tt*16+(lane&15);
    int nr=((lane&16)?8:0);
    for(int i=0;i<8;i++) sS[nr+i][m]=acc[i]*0.125f;
  }
  __syncthreads();
  // ---- I-BERT integer softmax, 16 threads per row ----
  float s8,z8; get_scale(scal,8,s8,z8);
  float x0=floorf(-0.69314718055994531f/s8);              // negative
  float bint=floorf(0.96963238f/(0.35815147f*s8));
  float cint=floorf(1.0f/(0.35815147f*s8*s8));
  int row=t>>4, l=t&15;
  float lmax=-3.0e38f;
  for(int c=l;c<197;c+=16){                               // fq_act(attn) fused: x_int = q - zero
    float xi=fminf(fmaxf(rintf(sS[row][c]/s8)+z8,0.0f),255.0f)-z8;
    sS[row][c]=xi; lmax=fmaxf(lmax,xi);
  }
  sred[row][l]=lmax; __syncthreads();
  if(l==0){ float m=sred[row][0]; for(int j=1;j<16;j++) m=fmaxf(m,sred[row][j]); sred[row][16]=m; }
  __syncthreads();
  float rmax=sred[row][16];
  float lsum=0.0f;
  for(int c=l;c<197;c+=16){
    float xi=fmaxf(sS[row][c]-rmax,30.0f*x0);
    float qe=floorf(xi/x0);
    float r=xi-x0*qe;
    float e=(r*(r+bint)+cint)*exp2f(-qe)*(0.35815147f*s8*s8);
    sS[row][c]=e; lsum+=e;
  }
  __syncthreads();
  sred[row][l]=lsum; __syncthreads();
  if(l==0){ float m=0.0f; for(int j=0;j<16;j++) m+=sred[row][j]; sred[row][16]=m; }
  __syncthreads();
  float inv=1.0f/fmaxf(sred[row][16],1e-30f);
  for(int c=l;c<224;c+=16)
    sP[row][c]=(c<197)?f2bf(sS[row][c]*inv):(unsigned short)0;
  __syncthreads();
  // ---- O = P[16x224] @ V[224x64]; waves 0..3 each own 16 output columns ----
  float fmn=3.0e38f, fmx=-3.0e38f;
  if(w<4){
    v8f acc={};
    for(int kk=0;kk<7;kk++){
      v16bf af=frag_a(&sP[lane&15][kk*32],lane);
      v16bf bfr=frag_b(&sVt[w*16+(lane&15)][kk*32],lane);
      acc=wmma_bf16(af,bfr,acc);
    }
    int d=h*64+w*16+(lane&15);
    int nb=mt*16+((lane&16)?8:0);
    for(int i=0;i<8;i++){
      int n=nb+i;
      if(n<197){
        float v=acc[i];
        fmn=fminf(fmn,v); fmx=fmaxf(fmx,v);
        AO[(size_t)(b*197+n)*768+d]=f2bf(v);
      }
    }
  }
  float2 mm=block_minmax(fmn,fmx);
  if(t==0){ atomicMin(&scal[10],f2ord(mm.x)); atomicMax(&scal[11],f2ord(mm.y)); }
}

// ---------------------------------------------------------------------------
extern "C" void kernel_launch(void* const* d_in, const int* in_sizes, int n_in,
                              void* d_out, int out_size, void* d_ws, size_t ws_size,
                              hipStream_t stream){
  (void)in_sizes; (void)n_in; (void)out_size; (void)ws_size;
  const float* x     =(const float*)d_in[0];
  const float* w_qkv =(const float*)d_in[1];
  const float* b_qkv =(const float*)d_in[2];
  const float* w_proj=(const float*)d_in[3];
  const float* b_proj=(const float*)d_in[4];
  float* out=(float*)d_out;

  const long long M=64ll*197;     // 12608 rows (multiple of 64)
  const int C=768, O=2304;

  char* ws=(char*)d_ws; size_t off=0;
  auto take=[&](size_t bytes)->char*{ char* p=ws+off; off+=(bytes+255)&~(size_t)255; return p; };
  unsigned*       scal  =(unsigned*)      take(256);
  unsigned short* Xb    =(unsigned short*)take((size_t)M*C*2);
  unsigned short* Wqkvb =(unsigned short*)take((size_t)O*C*2);
  unsigned short* Wprojb=(unsigned short*)take((size_t)C*C*2);
  unsigned short* QKVb  =(unsigned short*)take((size_t)M*O*2);
  unsigned short* AOb   =(unsigned short*)take((size_t)M*C*2);

  init_scal_kernel<<<1,64,0,stream>>>(scal);
  quant_w_kernel<<<O,256,0,stream>>>(w_qkv,Wqkvb,C);
  quant_w_kernel<<<C,256,0,stream>>>(w_proj,Wprojb,C);
  x2bf_kernel<<<4096,256,0,stream>>>(x,Xb,M*C);
  // qkv GEMM -> bf16 + global min/max (slots 0/1)
  gemm_bf16_kernel<<<dim3(O/128,(int)(M/64)),256,0,stream>>>(Xb,Wqkvb,b_qkv,QKVb,nullptr,
                                                             scal,0,2,768,O,C);
  // fq_act(qkv) globally, recording per-segment (q/k/v) min/max of the quantized values
  for(int s=0;s<3;s++)
    fq_apply_kernel<<<4096,256,0,stream>>>(QKVb,M,O,s*C,C,scal,0,&scal[2+2*s]);
  // second-stage per-segment fq of q / k / v
  for(int s=0;s<3;s++)
    fq_apply_kernel<<<4096,256,0,stream>>>(QKVb,M,O,s*C,C,scal,2+2*s,nullptr);
  attn_pass1_kernel<<<dim3(13,12,64),256,0,stream>>>(QKVb,scal);
  attn_pass2_kernel<<<dim3(13,12,64),256,0,stream>>>(QKVb,AOb,scal);
  fq_apply_kernel<<<4096,256,0,stream>>>(AOb,M,C,0,C,scal,10,nullptr);
  gemm_bf16_kernel<<<dim3(C/128,(int)(M/64)),256,0,stream>>>(AOb,Wprojb,b_proj,nullptr,out,
                                                             scal,12,0,0,C,C);
  fq_out_kernel<<<4096,256,0,stream>>>(out,M*C,scal,12);
}
